// attentionb_59124519797336
// MI455X (gfx1250) — matrix operand order repaired and checked by reading, not compiled, not run on hardware
//
#include <hip/hip_runtime.h>
#include <hip/hip_bf16.h>

#define BB 2
#define CC 1024
#define DD 1024
#define HH 16
#define HD 64
#define KQ 192      // augmented K-dim: [base | *Pe | *Ps] blocks of 64
#define SMPAD 1044  // f32 score row stride: 1044*4B -> lane lm hits bank lm*20 mod 64 (all distinct)

typedef __attribute__((ext_vector_type(16))) _Float16 v16h;
typedef __attribute__((ext_vector_type(8)))  float    v8f;
typedef __attribute__((ext_vector_type(4)))  unsigned int u32x4;

// ---- WMMA fragment loaders (wave32, v_wmma_f32_16x16x32_f16 layouts) ----
// A (16x32 f16): lane L -> M = L&15, group g = L>>4.
//   halves h=0..7  : K = k0 + g*8 + h          (contiguous 16B)
//   halves h=8..15 : K = k0 + 16 + g*8 + (h-8) (contiguous 16B)
static __device__ inline v16h load_a_frag(const _Float16* __restrict__ row, int k0, int half) {
  v16h a;
  ((u32x4*)&a)[0] = *(const u32x4*)(row + k0 + half * 8);
  ((u32x4*)&a)[1] = *(const u32x4*)(row + k0 + 16 + half * 8);
  return a;
}
// B (32x16 f16): lane L -> N = L&15, group g = L>>4; halves h=0..7 -> K = k0 + g*8 + h, h=8..15 -> K = k0 + 16 + g*8 + (h-8)
static __device__ inline v16h load_b_frag(const _Float16* __restrict__ row, int k0, int half) {
  v16h b;
  ((u32x4*)&b)[0] = *(const u32x4*)(row + k0 + half * 8);
  ((u32x4*)&b)[1] = *(const u32x4*)(row + k0 + 16 + half * 8);
  return b;
}

// ---- LayerNorm (one row per block), writes one or two f16 affine outputs ----
__global__ __launch_bounds__(256) void ln_dual_kernel(
    const float* __restrict__ in, const float* __restrict__ g1, const float* __restrict__ b1,
    const float* __restrict__ g2, const float* __restrict__ b2,
    _Float16* __restrict__ o1, _Float16* __restrict__ o2, int dual) {
  __shared__ float r1[256], r2[256];
  const int row = blockIdx.x, tid = threadIdx.x;
  const float* p = in + (size_t)row * DD;
  float s = 0.f, s2 = 0.f;
  for (int c = tid; c < DD; c += 256) { float v = p[c]; s += v; s2 += v * v; }
  r1[tid] = s; r2[tid] = s2; __syncthreads();
  for (int st = 128; st > 0; st >>= 1) {
    if (tid < st) { r1[tid] += r1[tid + st]; r2[tid] += r2[tid + st]; }
    __syncthreads();
  }
  const float mean = r1[0] * (1.f / DD);
  const float var  = r2[0] * (1.f / DD) - mean * mean;
  const float rstd = rsqrtf(var + 1e-5f);
  for (int c = tid; c < DD; c += 256) {
    float t = (p[c] - mean) * rstd;
    o1[(size_t)row * DD + c] = (_Float16)(t * g1[c] + b1[c]);
    if (dual) o2[(size_t)row * DD + c] = (_Float16)(t * g2[c] + b2[c]);
  }
}

// ---- Weight transpose + f32->f16: in[K,N] -> out[N,K] ----
__global__ __launch_bounds__(256) void transpose_cvt_kernel(
    const float* __restrict__ in, _Float16* __restrict__ out, int K, int N) {
  size_t idx = (size_t)blockIdx.x * 256 + threadIdx.x;
  if (idx >= (size_t)K * N) return;
  int k = (int)(idx / N), n = (int)(idx % N);
  out[(size_t)n * K + k] = (_Float16)in[idx];
}

// ---- WMMA GEMM: C[M,N] = A[M,K] @ Bt[N,K]^T + bias[N] ----
// 4 waves/block; each wave computes a 16x64 strip: one A-fragment feeds 4 WMMAs.
__global__ __launch_bounds__(128) void gemm_f16_nt(
    const _Float16* __restrict__ A, const _Float16* __restrict__ Bt,
    const float* __restrict__ bias, float* __restrict__ C, int M, int N, int K) {
  const int lane = threadIdx.x & 31, wid = threadIdx.x >> 5;
  const int lm = lane & 15, half = lane >> 4;
  const int m0 = blockIdx.x * 16;
  const int n0 = blockIdx.y * 256 + wid * 64;   // N must be a multiple of 256
  const _Float16* arow = A + (size_t)(m0 + lm) * K;
  const _Float16* brow = Bt + (size_t)(n0 + lm) * K;
  v8f acc0 = {}, acc1 = {}, acc2 = {}, acc3 = {};
  for (int k0 = 0; k0 < K; k0 += 32) {
    if (k0 + 256 < K) {   // gfx1250 global_prefetch_b8 for the next L2->L0 block
      __builtin_prefetch(arow + k0 + 256, 0, 1);
      __builtin_prefetch(brow + k0 + 256, 0, 1);
    }
    v16h a  = load_a_frag(arow, k0, half);
    v16h b0 = load_b_frag(brow,                     k0, half);
    v16h b1 = load_b_frag(brow + (size_t)16 * K,    k0, half);
    v16h b2 = load_b_frag(brow + (size_t)32 * K,    k0, half);
    v16h b3 = load_b_frag(brow + (size_t)48 * K,    k0, half);
    acc0 = __builtin_amdgcn_wmma_f32_16x16x32_f16(false, a, false, b0, (short)0, acc0, false, false);
    acc1 = __builtin_amdgcn_wmma_f32_16x16x32_f16(false, a, false, b1, (short)0, acc1, false, false);
    acc2 = __builtin_amdgcn_wmma_f32_16x16x32_f16(false, a, false, b2, (short)0, acc2, false, false);
    acc3 = __builtin_amdgcn_wmma_f32_16x16x32_f16(false, a, false, b3, (short)0, acc3, false, false);
  }
#pragma unroll
  for (int nt = 0; nt < 4; ++nt) {
    v8f acc = (nt == 0) ? acc0 : (nt == 1) ? acc1 : (nt == 2) ? acc2 : acc3;
    const int nc = n0 + nt * 16 + lm;
    const float bn = bias[nc];
#pragma unroll
    for (int r = 0; r < 8; ++r) {
      int m = m0 + r + half * 8;  // C/D layout: N=lane&15, M=r+8*group
      C[(size_t)m * N + nc] = acc[r] + bn;
    }
  }
}

// ---- Per-(b,h) banded prefix sums: Pe[i]=P[min(C,i+w)], Ps[i]=P[max(0,i-w)] ----
__global__ __launch_bounds__(64) void prefix_kernel(
    const float* __restrict__ ctx, const int* __restrict__ wp,
    float* __restrict__ Pe, float* __restrict__ Ps) {
  const int bh = blockIdx.x, d = threadIdx.x;
  const int b = bh >> 4, h = bh & 15;
  const int w = wp[0];
  const float* src = ctx + (size_t)b * CC * DD + (size_t)h * HD + d;
  float* pe = Pe + (size_t)bh * CC * HD + d;
  float* ps = Ps + (size_t)bh * CC * HD + d;
  for (int i = 0; i <= w && i < CC; ++i) ps[(size_t)i * HD] = 0.f;
  if (w == 0) pe[0] = 0.f;
  float s = 0.f;
  for (int t = 0; t < CC; ++t) {
    s += src[(size_t)t * DD];
    int i1 = t - w + 1;                     // Pe[i1] = sum_{u<=i1+w-1}
    if (i1 >= 0 && i1 < CC) pe[(size_t)i1 * HD] = s;
    int i2 = t + w + 1;                     // Ps[i2] = sum_{u<=i2-w-1}
    if (i2 < CC) ps[(size_t)i2 * HD] = s;
  }
  int start = CC - w + 1; if (start < 0) start = 0;
  for (int i = start; i < CC; ++i) pe[(size_t)i * HD] = s;  // clipped tail -> total
}

// ---- Build augmented 192-wide operands + V^T, all f16 ----
// S_low = [s*q, q*Pe, q] . [k, k, -k*Ps]   ;   S_up = [s*q, q, q*Ps] . [k, k*Pe, -k]
__global__ __launch_bounds__(256) void pack_kernel(
    const float* __restrict__ q, const float* __restrict__ kv,
    const float* __restrict__ Pe, const float* __restrict__ Ps,
    _Float16* __restrict__ QAl, _Float16* __restrict__ QAu,
    _Float16* __restrict__ KBl, _Float16* __restrict__ KBu, _Float16* __restrict__ Vt) {
  const unsigned idx = blockIdx.x * 256 + threadIdx.x;   // = bh*65536 + i*64 + d
  const int d = idx & 63;
  const int i = (idx >> 6) & 1023;
  const int bh = idx >> 16;
  const int b = bh >> 4, h = bh & 15;
  const float qv = q[((size_t)b * CC + i) * DD + h * HD + d];
  const float kk = kv[((size_t)b * CC + i) * (2 * DD) + h * HD + d];
  const float vv = kv[((size_t)b * CC + i) * (2 * DD) + DD + h * HD + d];
  const float pe = Pe[idx];
  const float ps = Ps[idx];
  const float scale = 0.125f;  // HD^-0.5
  const size_t ro = ((size_t)bh * CC + i) * KQ;
  QAl[ro + d]       = (_Float16)(scale * qv);
  QAl[ro + 64 + d]  = (_Float16)(qv * pe);
  QAl[ro + 128 + d] = (_Float16)qv;
  QAu[ro + d]       = (_Float16)(scale * qv);
  QAu[ro + 64 + d]  = (_Float16)qv;
  QAu[ro + 128 + d] = (_Float16)(qv * ps);
  KBl[ro + d]       = (_Float16)kk;
  KBl[ro + 64 + d]  = (_Float16)kk;
  KBl[ro + 128 + d] = (_Float16)(-kk * ps);
  KBu[ro + d]       = (_Float16)kk;
  KBu[ro + 64 + d]  = (_Float16)(kk * pe);
  KBu[ro + 128 + d] = (_Float16)(-kk);
  Vt[((size_t)bh * HD + d) * CC + i] = (_Float16)vv;
}

// ---- Attention: one wave per (bh, 16-row i-tile). Dynamic-LDS scores, softmax, AV. ----
__global__ __launch_bounds__(32) void attn_kernel(
    const _Float16* __restrict__ QAl, const _Float16* __restrict__ QAu,
    const _Float16* __restrict__ KBl, const _Float16* __restrict__ KBu,
    const _Float16* __restrict__ Vt, _Float16* __restrict__ wv) {
  extern __shared__ char smem_raw[];
  float (*sm)[SMPAD] = reinterpret_cast<float (*)[SMPAD]>(smem_raw);          // [row m][col j]
  float* sinv = reinterpret_cast<float*>(smem_raw + (size_t)16 * SMPAD * 4);  // [16] per-row 1/sum
  const int lane = threadIdx.x, lm = lane & 15, half = lane >> 4;
  const int i0 = blockIdx.x * 16;
  const int bh = blockIdx.y;
  const int b = bh >> 4, h = bh & 15;

  // A-fragments for both branches (K=192 -> 6 chunks of 32), reused across all 64 j-tiles
  v16h aL[6], aU[6];
  const _Float16* ql = QAl + ((size_t)bh * CC + i0 + lm) * KQ;
  const _Float16* qu = QAu + ((size_t)bh * CC + i0 + lm) * KQ;
#pragma unroll
  for (int kc = 0; kc < 6; ++kc) {
    aL[kc] = load_a_frag(ql, kc * 32, half);
    aU[kc] = load_a_frag(qu, kc * 32, half);
  }

  for (int jt = 0; jt < CC / 16; ++jt) {
    const int j0 = jt * 16;
    v8f cl = {}, cu = {};
    if (j0 <= i0) {  // tile contains i>=j elements
      const _Float16* kb = KBl + ((size_t)bh * CC + j0 + lm) * KQ;
#pragma unroll
      for (int kc = 0; kc < 6; ++kc) {
        v16h bf = load_b_frag(kb, kc * 32, half);
        cl = __builtin_amdgcn_wmma_f32_16x16x32_f16(false, aL[kc], false, bf, (short)0, cl, false, false);
      }
    }
    if (j0 >= i0) {  // tile contains i<j elements
      const _Float16* kb = KBu + ((size_t)bh * CC + j0 + lm) * KQ;
#pragma unroll
      for (int kc = 0; kc < 6; ++kc) {
        v16h bf = load_b_frag(kb, kc * 32, half);
        cu = __builtin_amdgcn_wmma_f32_16x16x32_f16(false, aU[kc], false, bf, (short)0, cu, false, false);
      }
    }
#pragma unroll
    for (int r = 0; r < 8; ++r) {
      int m = r + half * 8;
      int gi = i0 + m, gj = j0 + lm;
      sm[m][j0 + lm] = (gi >= gj) ? cl[r] : cu[r];
    }
  }
  __syncthreads();

  // softmax over j (vectorized): lane handles row lm, half-columns [half*512, +512)
  float* rowp = sm[lm] + half * 512;
  float mx = -3.0e38f;
  for (int c = 0; c < 512; c += 4) {
    float4 v = *(const float4*)(rowp + c);
    mx = fmaxf(mx, fmaxf(fmaxf(v.x, v.y), fmaxf(v.z, v.w)));
  }
  mx = fmaxf(mx, __shfl_xor(mx, 16));
  float sum = 0.f;
  for (int c = 0; c < 512; c += 4) {
    float4 v = *(const float4*)(rowp + c);
    v.x = __expf(v.x - mx); v.y = __expf(v.y - mx);
    v.z = __expf(v.z - mx); v.w = __expf(v.w - mx);
    sum += (v.x + v.y) + (v.z + v.w);
    *(float4*)(rowp + c) = v;   // store UNnormalized p
  }
  sum += __shfl_xor(sum, 16);
  if (half == 0) sinv[lm] = 1.f / sum;   // normalization deferred to the wv store
  __syncthreads();

  // AV: [16 x 1024] @ [1024 x 64] via WMMA; A rebuilt f32->f16 from LDS (4x ds_load_b128/frag)
  const _Float16* vb = Vt + (size_t)bh * HD * CC;
#pragma unroll 1
  for (int nt = 0; nt < 4; ++nt) {
    v8f acc = {};
    const _Float16* vrow = vb + (size_t)(nt * 16 + lm) * CC;
    const float* srow = sm[lm];
    for (int kc = 0; kc < CC / 32; ++kc) {
      const int kb = kc * 32;
      float4 r0 = *(const float4*)(srow + kb + half * 8);
      float4 r1 = *(const float4*)(srow + kb + half * 8 + 4);
      float4 r2 = *(const float4*)(srow + kb + 16 + half * 8);
      float4 r3 = *(const float4*)(srow + kb + 16 + half * 8 + 4);
      v16h a;
      _Float16* ap = (_Float16*)&a;
      ap[0] = (_Float16)r0.x;  ap[1] = (_Float16)r0.y;  ap[2] = (_Float16)r0.z;  ap[3] = (_Float16)r0.w;
      ap[4] = (_Float16)r1.x;  ap[5] = (_Float16)r1.y;  ap[6] = (_Float16)r1.z;  ap[7] = (_Float16)r1.w;
      ap[8] = (_Float16)r2.x;  ap[9] = (_Float16)r2.y;  ap[10] = (_Float16)r2.z; ap[11] = (_Float16)r2.w;
      ap[12] = (_Float16)r3.x; ap[13] = (_Float16)r3.y; ap[14] = (_Float16)r3.z; ap[15] = (_Float16)r3.w;
      v16h bf = load_b_frag(vrow, kb, half);
      acc = __builtin_amdgcn_wmma_f32_16x16x32_f16(false, a, false, bf, (short)0, acc, false, false);
    }
#pragma unroll
    for (int r = 0; r < 8; ++r) {
      int m = r + half * 8;
      wv[((size_t)b * CC + i0 + m) * DD + h * HD + nt * 16 + lm] = (_Float16)(acc[r] * sinv[m]);
    }
  }
}

extern "C" void kernel_launch(void* const* d_in, const int* in_sizes, int n_in,
                              void* d_out, int out_size, void* d_ws, size_t ws_size,
                              hipStream_t stream) {
  (void)in_sizes; (void)n_in; (void)out_size; (void)ws_size;
  const float* x   = (const float*)d_in[0];
  const float* pt  = (const float*)d_in[1];
  const float* gq  = (const float*)d_in[2];
  const float* bq  = (const float*)d_in[3];
  const float* Wq  = (const float*)d_in[4];
  const float* cq  = (const float*)d_in[5];
  const float* gc  = (const float*)d_in[6];
  const float* bc  = (const float*)d_in[7];
  const float* Wc  = (const float*)d_in[8];
  const float* cc  = (const float*)d_in[9];
  const float* gkv = (const float*)d_in[10];
  const float* bkv = (const float*)d_in[11];
  const float* Wkv = (const float*)d_in[12];
  const float* ckv = (const float*)d_in[13];
  const float* Wo  = (const float*)d_in[14];
  const float* co  = (const float*)d_in[15];
  const int*   wp  = (const int*)d_in[16];
  float* out = (float*)d_out;

  const size_t MB = 1u << 20;
  char* ws = (char*)d_ws;
  size_t off = 0;
  auto take = [&](size_t bytes) { char* p = ws + off; off += (bytes + 255) & ~(size_t)255; return p; };

  // Region A (48MB): early = LN outputs + transposed weights; late = augmented Q'/K'
  char* regA = take(48 * MB);
  _Float16* xq   = (_Float16*)(regA + 0 * MB);   // 4MB
  _Float16* xkv  = (_Float16*)(regA + 4 * MB);   // 4MB
  _Float16* pc   = (_Float16*)(regA + 8 * MB);   // 4MB
  _Float16* Wqt  = (_Float16*)(regA + 12 * MB);  // 2MB
  _Float16* Wct  = (_Float16*)(regA + 14 * MB);  // 2MB
  _Float16* Wkvt = (_Float16*)(regA + 16 * MB);  // 4MB
  _Float16* QAl  = (_Float16*)(regA + 0 * MB);   // 12MB (aliases dead after GEMMs)
  _Float16* QAu  = (_Float16*)(regA + 12 * MB);  // 12MB
  _Float16* KBl  = (_Float16*)(regA + 24 * MB);  // 12MB
  _Float16* KBu  = (_Float16*)(regA + 36 * MB);  // 12MB

  float* qf   = (float*)take(8 * MB);
  float* kvf  = (float*)take(16 * MB);
  float* ctxf = (float*)take(8 * MB);
  char* regE = take(16 * MB);                    // Pe+Ps early; wv late
  float* Pe = (float*)regE;
  float* Ps = (float*)(regE + 8 * MB);
  _Float16* wvh = (_Float16*)regE;               // 4MB, after Pe/Ps are dead
  _Float16* Vt  = (_Float16*)take(4 * MB);
  _Float16* Wot = (_Float16*)take(2 * MB);

  // 1. LayerNorms -> f16
  ln_dual_kernel<<<BB * CC, 256, 0, stream>>>(x, gq, bq, gkv, bkv, xq, xkv, 1);
  ln_dual_kernel<<<BB * CC, 256, 0, stream>>>(pt, gc, bc, gc, bc, pc, pc, 0);
  // 2. Weights -> transposed f16
  transpose_cvt_kernel<<<(DD * DD) / 256, 256, 0, stream>>>(Wq, Wqt, DD, DD);
  transpose_cvt_kernel<<<(DD * DD) / 256, 256, 0, stream>>>(Wc, Wct, DD, DD);
  transpose_cvt_kernel<<<(DD * 2 * DD) / 256, 256, 0, stream>>>(Wkv, Wkvt, DD, 2 * DD);
  transpose_cvt_kernel<<<(DD * DD) / 256, 256, 0, stream>>>(Wo, Wot, DD, DD);
  // 3. Projection GEMMs (WMMA, 16x64 per wave)
  gemm_f16_nt<<<dim3((BB * CC) / 16, DD / 256), 128, 0, stream>>>(xq, Wqt, cq, qf, BB * CC, DD, DD);
  gemm_f16_nt<<<dim3((BB * CC) / 16, DD / 256), 128, 0, stream>>>(pc, Wct, cc, ctxf, BB * CC, DD, DD);
  gemm_f16_nt<<<dim3((BB * CC) / 16, (2 * DD) / 256), 128, 0, stream>>>(xkv, Wkvt, ckv, kvf, BB * CC, 2 * DD, DD);
  // 4. Banded prefix sums, 5. operand packing
  prefix_kernel<<<BB * HH, HD, 0, stream>>>(ctxf, wp, Pe, Ps);
  pack_kernel<<<(BB * HH * CC * HD) / 256, 256, 0, stream>>>(qf, kvf, Pe, Ps, QAl, QAu, KBl, KBu, Vt);
  // 6. Attention (WMMA scores + softmax + WMMA AV); dynamic LDS: 16*SMPAD*4 + 64
  const size_t attn_lds = (size_t)16 * SMPAD * sizeof(float) + 16 * sizeof(float);
  attn_kernel<<<dim3(CC / 16, BB * HH), 32, attn_lds, stream>>>(QAl, QAu, KBl, KBu, Vt, wvh);
  // 7. Output projection (WMMA), writes f32 d_out
  gemm_f16_nt<<<dim3((BB * CC) / 16, DD / 256), 128, 0, stream>>>(wvh, Wot, co, out, BB * CC, DD, DD);
}
